// ReweightGNN_11081015624122
// MI455X (gfx1250) — compile-verified, matrix operand order checked by reading, not run on hardware
//
#include <hip/hip_runtime.h>

// ---------------------------------------------------------------------------
// ReweightGNN (GraphSAGE w/ edge reweighting) for MI455X / gfx1250.
// GEMMs via v_wmma_f32_16x16x32_bf16; scatter via global f32 atomics.
// ---------------------------------------------------------------------------

typedef __attribute__((ext_vector_type(16))) __bf16 v16bf;
typedef __attribute__((ext_vector_type(8)))  float  v8f;

#define DG   128
#define DOUT 10
#define LMDA 1.0f

// ---------------- helpers ----------------

__device__ __forceinline__ unsigned short f2bf(float f) {
    union { float f; unsigned int u; } c; c.f = f;
    unsigned int u = c.u;
    u += 0x7fffu + ((u >> 16) & 1u);            // round-to-nearest-even
    return (unsigned short)(u >> 16);
}

__device__ __forceinline__ unsigned int hash32(unsigned int x) {
    x ^= x >> 16; x *= 0x7feb352dU;
    x ^= x >> 15; x *= 0x846ca68bU;
    x ^= x >> 16; return x;
}

// Load one 16x32 bf16 WMMA fragment (A-layout; B uses same layout on W^T).
// Layout per CDNA5 ISA §7.12.2 (16-bit A 16x32):
//   lane r=lane&15 -> matrix row; hs=lane>>4 selects K-half.
// The 8 dword loads are contiguous per half, so the backend fuses them into
// global_load_b128 clauses.
__device__ __forceinline__ v16bf load_frag(const unsigned short* rowptr, int hs) {
    union { v16bf v; unsigned int u[8]; } t;
    const unsigned short* p = rowptr + 8 * hs;
#pragma unroll
    for (int i = 0; i < 8; ++i) {
        int off = 2 * i + ((i >= 4) ? 8 : 0);   // K pair offset within half
        t.u[i] = *(const unsigned int*)(p + off);
    }
    return t.v;
}

// ---------------- GEMM: C[nrows x ncols] = A[nrows x K] @ Bt^T + bias --------
// A: bf16 row-major (nrows x K).  Bt: bf16 row-major (ncols x K) == W^T.
// NK = K/32 (A-fragments live in registers, K-loop unrolls).
// GUARD=false -> unconditional stores (requires nrows%16==0, store_cols==ncols).

template <int NK, bool GUARD, bool RELU>
__global__ __launch_bounds__(256)
void gemm_bias_wmma(const unsigned short* __restrict__ A,
                    const unsigned short* __restrict__ Bt,
                    const float* __restrict__ bias,
                    float* __restrict__ C,
                    int nrows, int ncols, int ldc, int store_cols)
{
    const int K = NK * 32;
    int wave = blockIdx.x * (blockDim.x >> 5) + (threadIdx.x >> 5);
    int row0 = wave * 16;
    if (row0 >= nrows) return;                  // wave-uniform exit

    int lane = threadIdx.x & 31;
    int r  = lane & 15;
    int hs = lane >> 4;

    int ar = row0 + r; if (ar > nrows - 1) ar = nrows - 1;   // row clamp
    const unsigned short* arow = A + (size_t)ar * K;

    v16bf afrag[NK];
#pragma unroll
    for (int kk = 0; kk < NK; ++kk) afrag[kk] = load_frag(arow + kk * 32, hs);

    for (int n0 = 0; n0 < ncols; n0 += 16) {
        const unsigned short* brow = Bt + (size_t)(n0 + r) * K;
        v8f acc = {};
#pragma unroll
        for (int kk = 0; kk < NK; ++kk) {
            v16bf bfrag = load_frag(brow + kk * 32, hs);
            acc = __builtin_amdgcn_wmma_f32_16x16x32_bf16(
                      false, afrag[kk], false, bfrag,
                      (short)0, acc, false, false);
        }
        int col = n0 + r;                        // C layout: N = lane&15
        if (GUARD) {
            float bv = (col < store_cols) ? bias[col] : 0.0f;
#pragma unroll
            for (int m = 0; m < 8; ++m) {        // C layout: M = m + 8*hs
                int rr = row0 + m + 8 * hs;
                float val = acc[m] + bv;
                if (RELU) val = fmaxf(val, 0.0f);
                if (rr < nrows && col < store_cols)
                    C[(size_t)rr * ldc + col] = val;
            }
        } else {
            float bv = bias[col];
            float* cptr = C + (size_t)(row0 + 8 * hs) * ldc + col;
#pragma unroll
            for (int m = 0; m < 8; ++m) {
                float val = acc[m] + bv;
                if (RELU) val = fmaxf(val, 0.0f);
                cptr[(size_t)m * ldc] = val;
            }
        }
    }
}

// ---------------- small utility kernels ----------------

__global__ void fill_f32(float* p, float v, size_t n) {
    size_t i = (size_t)blockIdx.x * blockDim.x + threadIdx.x;
    if (i < n) p[i] = v;
}

__global__ void conv_bf16(const float* __restrict__ in,
                          unsigned short* __restrict__ out, size_t n) {
    size_t i = (size_t)blockIdx.x * blockDim.x + threadIdx.x;
    if (i < n) out[i] = f2bf(in[i]);
}

// W (K x ncols, f32 row-major) -> Wt (npad x K, bf16 row-major), zero-padded cols.
__global__ void wconv_t(const float* __restrict__ W, unsigned short* __restrict__ Wt,
                        int K, int ncols, int npad) {
    int idx = blockIdx.x * blockDim.x + threadIdx.x;
    int total = K * npad;
    if (idx >= total) return;
    int k = idx / npad, n = idx - k * npad;
    float v = (n < ncols) ? W[(size_t)k * ncols + n] : 0.0f;
    Wt[(size_t)n * K + k] = f2bf(v);
}

__global__ void deg_kernel(const long long* __restrict__ ei, float* cnt, int E) {
    int e = blockIdx.x * blockDim.x + threadIdx.x;
    if (e < E) atomicAdd(&cnt[(int)ei[e]], 1.0f);      // row = edge_index[0][e]
}

__global__ void inv_kernel(const float* __restrict__ cnt, float* inv, int N) {
    int i = blockIdx.x * blockDim.x + threadIdx.x;
    if (i < N) inv[i] = 1.0f / fmaxf(cnt[i], 1.0f);
}

// One wave per edge: gather H[col] (float4/lane), scale, atomic-add into AGG[row].
// Edge id is forced wave-uniform so ei/ew reads become scalar (SMEM) loads.
__global__ __launch_bounds__(256)
void edge_scatter(const long long* __restrict__ ei, const float* __restrict__ ew,
                  const float* __restrict__ H, float* __restrict__ agg, int E) {
    int e = blockIdx.x * (blockDim.x >> 5) + (threadIdx.x >> 5);
    e = __builtin_amdgcn_readfirstlane(e);
    if (e >= E) return;
    int lane = threadIdx.x & 31;
    int row = (int)ei[e];
    int col = (int)ei[(size_t)E + e];
    float w = (1.0f - LMDA) + LMDA * ew[e];
    const float4 v = *(const float4*)(H + (size_t)col * DG + lane * 4);
    float* dst = agg + (size_t)row * DG + lane * 4;
    atomicAdd(dst + 0, v.x * w);
    atomicAdd(dst + 1, v.y * w);
    atomicAdd(dst + 2, v.z * w);
    atomicAdd(dst + 3, v.w * w);
}

// Build bf16 concat input [agg*inv_cnt | x_in]  (N x 256).
__global__ void concat_kernel(const float* __restrict__ agg,
                              const float* __restrict__ invc,
                              const float* __restrict__ xin,
                              unsigned short* __restrict__ out, int N) {
    size_t idx = (size_t)blockIdx.x * blockDim.x + threadIdx.x;
    if (idx >= (size_t)N * 256) return;
    int n = (int)(idx >> 8), c = (int)(idx & 255);
    float v = (c < DG) ? agg[(size_t)n * DG + c] * invc[n]
                       : xin[(size_t)n * DG + (c - DG)];
    out[idx] = f2bf(v);
}

// Deterministic dropout (p=0.5, scale 2.0), in place.
__global__ void dropout_kernel(float* __restrict__ h, size_t n, unsigned int seed) {
    size_t i = (size_t)blockIdx.x * blockDim.x + threadIdx.x;
    if (i >= n) return;
    unsigned int r = hash32((unsigned int)i * 0x9E3779B9u + seed);
    h[i] = (r & 1u) ? h[i] * 2.0f : 0.0f;
}

// ---------------- host-side orchestration ----------------

static inline int cdiv(long long a, long long b) { return (int)((a + b - 1) / b); }

extern "C" void kernel_launch(void* const* d_in, const int* in_sizes, int n_in,
                              void* d_out, int out_size, void* d_ws, size_t ws_size,
                              hipStream_t stream) {
    const float*     x        = (const float*)d_in[0];
    const long long* ei       = (const long long*)d_in[1];   // int64 (2,E)
    const float*     ew       = (const float*)d_in[2];
    const float*     W_lin_in = (const float*)d_in[3];
    const float*     b_lin_in = (const float*)d_in[4];
    const float*     W_agg_in = (const float*)d_in[5];
    const float*     b_agg_in = (const float*)d_in[6];
    const float*     W_lin_h  = (const float*)d_in[7];
    const float*     b_lin_h  = (const float*)d_in[8];
    const float*     W_agg_h  = (const float*)d_in[9];
    const float*     b_agg_h  = (const float*)d_in[10];
    const float*     W_mlp1   = (const float*)d_in[11];
    const float*     b_mlp1   = (const float*)d_in[12];
    const float*     W_mlp2   = (const float*)d_in[13];
    const float*     b_mlp2   = (const float*)d_in[14];

    const int N = in_sizes[0] / DG;
    const int E = in_sizes[2];
    const size_t NB = (size_t)N * DG;           // elems in an N x 128 matrix

    // Bump-allocate workspace (256B aligned).
    char* wsp = (char*)d_ws;
    auto alloc = [&](size_t bytes) -> void* {
        void* p = (void*)wsp;
        wsp += (bytes + 255) & ~(size_t)255;
        return p;
    };
    float*          Hlin = (float*)alloc(NB * 4);             // lin(x) output / MLP z
    float*          AGG  = (float*)alloc(NB * 4);             // scatter accumulator
    unsigned short* XB1  = (unsigned short*)alloc(NB * 2);    // bf16 A, K=128
    unsigned short* XB2  = (unsigned short*)alloc(NB * 2 * 2);// bf16 A, K=256
    float*          cnt  = (float*)alloc((size_t)N * 4);
    float*          invc = (float*)alloc((size_t)N * 4);
    unsigned short* WtLI = (unsigned short*)alloc((size_t)DG * DG * 2);
    unsigned short* WtAI = (unsigned short*)alloc((size_t)DG * 2 * DG * 2);
    unsigned short* WtLH = (unsigned short*)alloc((size_t)DG * DG * 2);
    unsigned short* WtAH = (unsigned short*)alloc((size_t)DG * 2 * DG * 2);
    unsigned short* WtM1 = (unsigned short*)alloc((size_t)DG * DG * 2);
    unsigned short* WtM2 = (unsigned short*)alloc((size_t)16 * DG * 2);

    float* hcur = (float*)d_out;                // running hidden state -> output h
    float* yout = (float*)d_out + NB;           // output y (N x 10)

    const int T = 256;

    // --- weight transpose+convert to bf16 (W^T, ncols x K row-major) ---
    wconv_t<<<cdiv(DG * DG, T), T, 0, stream>>>(W_lin_in, WtLI, DG, DG, DG);
    wconv_t<<<cdiv(2 * DG * DG, T), T, 0, stream>>>(W_agg_in, WtAI, 2 * DG, DG, DG);
    wconv_t<<<cdiv(DG * DG, T), T, 0, stream>>>(W_lin_h, WtLH, DG, DG, DG);
    wconv_t<<<cdiv(2 * DG * DG, T), T, 0, stream>>>(W_agg_h, WtAH, 2 * DG, DG, DG);
    wconv_t<<<cdiv(DG * DG, T), T, 0, stream>>>(W_mlp1, WtM1, DG, DG, DG);
    wconv_t<<<cdiv(DG * 16, T), T, 0, stream>>>(W_mlp2, WtM2, DG, DOUT, 16);

    // --- degree / inv_cnt ---
    fill_f32<<<cdiv(N, T), T, 0, stream>>>(cnt, 0.0f, (size_t)N);
    deg_kernel<<<cdiv(E, T), T, 0, stream>>>(ei, cnt, E);
    inv_kernel<<<cdiv(N, T), T, 0, stream>>>(cnt, invc, N);

    const int gemm_blocks = cdiv(cdiv(N, 16), 8);   // 8 waves / 256-thread block
    const int edge_blocks = cdiv(E, 8);             // 8 edges (waves) / block

    // --- one GNN layer ---
    auto layer = [&](const float* In, const unsigned short* WtL, const float* bL,
                     const unsigned short* WtA, const float* bA,
                     float* Out, unsigned int seed) {
        conv_bf16<<<cdiv((long long)NB, T), T, 0, stream>>>(In, XB1, NB);
        gemm_bias_wmma<4, false, false><<<gemm_blocks, T, 0, stream>>>(
            XB1, WtL, bL, Hlin, N, DG, DG, DG);
        fill_f32<<<cdiv((long long)NB, T), T, 0, stream>>>(AGG, 0.0f, NB);
        edge_scatter<<<edge_blocks, T, 0, stream>>>(ei, ew, Hlin, AGG, E);
        concat_kernel<<<cdiv((long long)NB * 2, T), T, 0, stream>>>(AGG, invc, In, XB2, N);
        gemm_bias_wmma<8, false, true><<<gemm_blocks, T, 0, stream>>>(
            XB2, WtA, bA, Out, N, DG, DG, DG);
        dropout_kernel<<<cdiv((long long)NB, T), T, 0, stream>>>(Out, NB, seed);
    };

    layer(x,    WtLI, b_lin_in, WtAI, b_agg_in, hcur, 0x0001u);
    layer(hcur, WtLH, b_lin_h,  WtAH, b_agg_h,  hcur, 0x0002u);
    layer(hcur, WtLH, b_lin_h,  WtAH, b_agg_h,  hcur, 0x0003u);

    // --- MLP head: y = relu(h @ W1 + b1) @ W2 + b2 ---
    conv_bf16<<<cdiv((long long)NB, T), T, 0, stream>>>(hcur, XB1, NB);
    gemm_bias_wmma<4, false, true><<<gemm_blocks, T, 0, stream>>>(
        XB1, WtM1, b_mlp1, Hlin, N, DG, DG, DG);
    conv_bf16<<<cdiv((long long)NB, T), T, 0, stream>>>(Hlin, XB1, NB);
    gemm_bias_wmma<4, true, false><<<gemm_blocks, T, 0, stream>>>(
        XB1, WtM2, b_mlp2, yout, N, 16, DOUT, DOUT);
    (void)n_in; (void)out_size; (void)ws_size;
}
